// Head_71339406787184
// MI455X (gfx1250) — matrix operand order repaired and checked by reading, not compiled, not run on hardware
//
#include <hip/hip_runtime.h>
#include <stdint.h>

typedef __bf16 bft;
typedef __attribute__((ext_vector_type(16))) __bf16 v16bf;
typedef __attribute__((ext_vector_type(8)))  __bf16 v8bf;
typedef __attribute__((ext_vector_type(8)))  float  v8f;
typedef __attribute__((ext_vector_type(4)))  unsigned int u32x4;
typedef __attribute__((ext_vector_type(8)))  int i32x8;
typedef __attribute__((ext_vector_type(4)))  int i32x4;

#define BT      16384      // B*T
#define TSEQ    2048
#define CDIM    1024
#define HEADS   64
#define NEG_BIG (-1.0e30f)

// ---------------------------------------------------------------------------
// Kernel 0: transpose + convert Wk|Wq|Wv (C x 64 f32) -> WT (3 x 64 x C bf16)
// WT is a contiguous (192, 1024) bf16 matrix -> a perfect 2-D TDM tensor.
// ---------------------------------------------------------------------------
__global__ __launch_bounds__(256) void wt_convert(const float* __restrict__ Wk,
                                                  const float* __restrict__ Wq,
                                                  const float* __restrict__ Wv,
                                                  bft* __restrict__ WT) {
  int i = blockIdx.x * blockDim.x + threadIdx.x;   // [0, 3*64*1024)
  if (i >= 3 * HEADS * CDIM) return;
  int m = i >> 16;
  int rem = i & 65535;
  int h = rem >> 10;            // head dim
  int c = rem & 1023;           // C dim
  const float* W = (m == 0) ? Wk : ((m == 1) ? Wq : Wv);
  WT[i] = (bft)W[(size_t)c * HEADS + h];
}

// ---------------------------------------------------------------------------
// Kernel 1: q,k,v projections. One wave per 16-row tile of x.
//   W slice (192 rows x 32 bf16) staged in LDS via TENSOR_LOAD_TO_LDS (TDM),
//   double-buffered; TDM pad fields reproduce the WPAD=40 bank-padded layout.
// ---------------------------------------------------------------------------
#define WPAD 40                         // padded LDS row stride (elems)
#define WBUF (12 * 16 * WPAD)           // 7680 elems = 15360 bytes per buffer

__global__ __launch_bounds__(256) void qkv_proj(const float* __restrict__ x,
                                                const bft* __restrict__ WT,
                                                bft* __restrict__ kout,
                                                bft* __restrict__ qout,
                                                bft* __restrict__ vT) {
  __shared__ bft ldsW[2 * WBUF];
  const int tid   = threadIdx.x;
  const int lane  = tid & 31;
  const int wslot = tid >> 5;
  const int wid   = blockIdx.x * 8 + wslot;   // [0, 1024)
  const int row0  = wid * 16;                 // global x-row tile base
  const int n     = lane & 15;
  const int hi    = lane >> 4;

  const float* xr = x + (size_t)(row0 + n) * CDIM;

  const uint32_t lds0 = (uint32_t)(uintptr_t)(&ldsW[0]);  // flat->LDS: low 32 bits
  const uint64_t wt0  = (uint64_t)(uintptr_t)WT;

  // TDM descriptor (ISA 8.3/8.4): 2-D tile, 192 rows x 32 bf16, row stride 1024,
  // LDS pad: 16 dwords data then 4 dwords pad -> 80-byte rows (= WPAD elems).
  auto tdm_issue = [&](int buf, int kb) {
    uint64_t ga = wt0 + (uint64_t)kb * 2;
    u32x4 g0;
    g0[0] = 1u;                                              // count=1 (valid)
    g0[1] = lds0 + (uint32_t)buf * (uint32_t)(WBUF * 2);     // lds_addr (bytes)
    g0[2] = (uint32_t)ga;                                    // global_addr lo
    g0[3] = (uint32_t)((ga >> 32) & 0x01FFFFFFu) | (2u << 30); // ga hi | type=2
    i32x8 g1;
    g1[0] = (int)((1u << 16) | (1u << 20) | (3u << 22) | (3u << 25));
            // data_size=2B | pad_enable | pad_interval=16dw | pad_amount=4dw
    g1[1] = (int)((uint32_t)CDIM << 16);   // tensor_dim0 = 1024 (bits 79:48 lo)
    g1[2] = (int)(192u << 16);             // tensor_dim1 = 192  (bits 111:80 lo)
    g1[3] = (int)(32u << 16);              // tile_dim0 = 32     (bits 127:112)
    g1[4] = 192;                           // tile_dim1 = 192    (bits 143:128)
    g1[5] = CDIM;                          // tensor_dim0_stride = 1024 (lo 32)
    g1[6] = 0;
    g1[7] = 0;
    i32x4 gz4 = {0, 0, 0, 0};
    i32x8 gz8 = {0, 0, 0, 0, 0, 0, 0, 0};
    __builtin_amdgcn_tensor_load_to_lds(g0, g1, gz4, gz4, gz8, 0);
  };

  const v8f vzero = {};
  v8f acc[12];
#pragma unroll
  for (int i = 0; i < 12; ++i) acc[i] = vzero;

  if (wslot == 0) tdm_issue(0, 0);
  int cur = 0;

  for (int kb = 0; kb < CDIM; kb += 32) {
    __builtin_amdgcn_s_wait_tensorcnt(0);     // issuing wave: current buf ready
    __syncthreads();                          // buf visible; prev reads done
    if (wslot == 0 && kb + 32 < CDIM) tdm_issue(cur ^ 1, kb + 32);

    // A fragment: 16x32 f32 tile of x -> bf16 (ISA A-layout)
    float4 f0 = *(const float4*)(xr + kb + hi * 8);
    float4 f1 = *(const float4*)(xr + kb + hi * 8 + 4);
    float4 f2 = *(const float4*)(xr + kb + 16 + hi * 8);
    float4 f3 = *(const float4*)(xr + kb + 16 + hi * 8 + 4);
    v16bf a;
    a[0]=(bft)f0.x; a[1]=(bft)f0.y; a[2]=(bft)f0.z; a[3]=(bft)f0.w;
    a[4]=(bft)f1.x; a[5]=(bft)f1.y; a[6]=(bft)f1.z; a[7]=(bft)f1.w;
    a[8]=(bft)f2.x; a[9]=(bft)f2.y; a[10]=(bft)f2.z; a[11]=(bft)f2.w;
    a[12]=(bft)f3.x; a[13]=(bft)f3.y; a[14]=(bft)f3.z; a[15]=(bft)f3.w;

    const bft* Wb = ldsW + cur * WBUF;

    // software-pipelined B fragments: load tile t+1 while WMMA on tile t
    union BU { v16bf v; v8bf h[2]; };
    BU bcur, bnxt;
    {
      const bft* bp = Wb + (0 * 16 + n) * WPAD + hi * 16;
      bcur.h[0] = *(const v8bf*)(bp - hi * 8);
      bcur.h[1] = *(const v8bf*)(bp - hi * 8 + 16);
    }
#pragma unroll
    for (int t = 0; t < 12; ++t) {
      if (t < 11) {
        const bft* bp = Wb + ((t + 1) * 16 + n) * WPAD + hi * 16;
        bnxt.h[0] = *(const v8bf*)(bp - hi * 8);
        bnxt.h[1] = *(const v8bf*)(bp - hi * 8 + 16);
      }
      acc[t] = __builtin_amdgcn_wmma_f32_16x16x32_bf16(
          false, a, false, bcur.v, (short)0, acc[t], false, false);
      bcur = bnxt;
    }
    cur ^= 1;
  }

  // C/D layout: VGPR r -> row (r + 8*hi), col = lane&15 within n-tile
#pragma unroll
  for (int nt = 0; nt < 4; ++nt) {
    int col = nt * 16 + n;
    v8f ak = acc[nt];       // m=0 -> Wk
    v8f aq = acc[4 + nt];   // m=1 -> Wq
    v8f av = acc[8 + nt];   // m=2 -> Wv
#pragma unroll
    for (int r = 0; r < 8; ++r) {
      size_t row = (size_t)(row0 + r + 8 * hi);
      kout[row * HEADS + col] = (bft)ak[r];
      qout[row * HEADS + col] = (bft)aq[r];
    }
    v8bf pv;
#pragma unroll
    for (int r = 0; r < 8; ++r) pv[r] = (bft)av[r];
    *(v8bf*)(vT + (size_t)col * BT + row0 + 8 * hi) = pv;   // vT: (64, BT)
  }
}

// ---------------------------------------------------------------------------
// Kernel 2: causal flash attention. One wave per (batch, 16-query tile).
// ---------------------------------------------------------------------------
#define PPAD 40
__global__ __launch_bounds__(256) void attn(const bft* __restrict__ qb,
                                            const bft* __restrict__ kbuf,
                                            const bft* __restrict__ vT,
                                            float* __restrict__ out) {
  __shared__ bft lds_p[8][16 * PPAD];
  const int lane  = threadIdx.x & 31;
  const int wslot = threadIdx.x >> 5;
  const int wid   = blockIdx.x * 8 + wslot;   // [0, 1024)
  const int b     = wid >> 7;                 // batch
  const int qrow0 = (wid & 127) * 16;         // query tile base within batch
  const int grow0 = b * TSEQ + qrow0;
  const int n     = lane & 15;
  const int hi    = lane >> 4;

  bft* pl = &lds_p[wslot][0];

  // q A-fragments for the two 32-wide head-dim K slices (kept in regs)
  v16bf aQ[2];
  {
    const bft* qr = qb + (size_t)(grow0 + n) * HEADS;
#pragma unroll
    for (int s = 0; s < 2; ++s) {
      union { v16bf v; v8bf h[2]; } u;
      u.h[0] = *(const v8bf*)(qr + s * 32 + hi * 16);
      u.h[1] = *(const v8bf*)(qr + s * 32 + hi * 16 + 8);
      aQ[s] = u.v;
    }
  }

  const v8f vzero = {};
  v8f o[4];
#pragma unroll
  for (int t = 0; t < 4; ++t) o[t] = vzero;
  float mrow[8], lrow[8];
#pragma unroll
  for (int r = 0; r < 8; ++r) { mrow[r] = NEG_BIG; lrow[r] = 0.0f; }

  for (int kb = 0; kb < qrow0 + 16; kb += 32) {
    // ---- S = q @ k^T : two 16x16 f32 tiles (keys kb..kb+15, kb+16..kb+31)
    v8f s[2];
#pragma unroll
    for (int t2 = 0; t2 < 2; ++t2) {
      const bft* kr = kbuf + (size_t)(b * TSEQ + kb + t2 * 16 + n) * HEADS;
      v16bf b0 = *(const v16bf*)(kr + hi * 16);        // head dims 0..31
      v16bf b1 = *(const v16bf*)(kr + 32 + hi * 16);   // head dims 32..63
      v8f c = vzero;
      c = __builtin_amdgcn_wmma_f32_16x16x32_bf16(false, aQ[0], false, b0, (short)0, c, false, false);
      c = __builtin_amdgcn_wmma_f32_16x16x32_bf16(false, aQ[1], false, b1, (short)0, c, false, false);
      s[t2] = c;
    }

    // ---- scale, causal mask, online softmax (rows live in 16-lane halves)
    float p0[8], p1[8];
#pragma unroll
    for (int r = 0; r < 8; ++r) {
      int qrow = qrow0 + r + 8 * hi;
      float v0 = s[0][r] * 0.125f;              // 1/sqrt(64)
      float v1 = s[1][r] * 0.125f;
      if (kb + n      > qrow) v0 = NEG_BIG;
      if (kb + 16 + n > qrow) v1 = NEG_BIG;
      float t = fmaxf(v0, v1);
      t = fmaxf(t, __shfl_xor(t, 8));
      t = fmaxf(t, __shfl_xor(t, 4));
      t = fmaxf(t, __shfl_xor(t, 2));
      t = fmaxf(t, __shfl_xor(t, 1));
      float mnew = fmaxf(mrow[r], t);
      float so   = __expf(mrow[r] - mnew);
      float e0   = __expf(v0 - mnew);
      float e1   = __expf(v1 - mnew);
      float rs = e0 + e1;
      rs += __shfl_xor(rs, 8);
      rs += __shfl_xor(rs, 4);
      rs += __shfl_xor(rs, 2);
      rs += __shfl_xor(rs, 1);
      lrow[r] = lrow[r] * so + rs;
      mrow[r] = mnew;
      p0[r] = e0; p1[r] = e1;
      o[0][r] *= so; o[1][r] *= so; o[2][r] *= so; o[3][r] *= so;
    }

    // ---- C-layout P -> row-major bf16 in LDS (per-wave region)
#pragma unroll
    for (int r = 0; r < 8; ++r) {
      int prow = r + 8 * hi;
      pl[prow * PPAD + n]      = (bft)p0[r];
      pl[prow * PPAD + 16 + n] = (bft)p1[r];
    }
    asm volatile("s_wait_dscnt 0" ::: "memory");   // cross-lane LDS RAW, wave-local

    // ---- re-read P in A-layout
    v16bf aP;
    {
      union { v16bf v; v8bf h[2]; } u;
      u.h[0] = *(const v8bf*)(pl + n * PPAD + hi * 16);
      u.h[1] = *(const v8bf*)(pl + n * PPAD + hi * 16 + 8);
      aP = u.v;
    }

    // ---- o += P @ V  (v^T gives contiguous per-lane B fragments)
#pragma unroll
    for (int t = 0; t < 4; ++t) {
      const bft* vr = vT + (size_t)(t * 16 + n) * BT + b * TSEQ + kb + hi * 16;
      v16bf bv = *(const v16bf*)vr;
      o[t] = __builtin_amdgcn_wmma_f32_16x16x32_bf16(
          false, aP, false, bv, (short)0, o[t], false, false);
    }
  }

  // ---- normalize and store fp32 output (B,T,64)
#pragma unroll
  for (int r = 0; r < 8; ++r) {
    float inv = 1.0f / lrow[r];
    size_t row = (size_t)(grow0 + r + 8 * hi);
#pragma unroll
    for (int t = 0; t < 4; ++t) {
      out[row * HEADS + t * 16 + n] = o[t][r] * inv;
    }
  }
}

// ---------------------------------------------------------------------------
extern "C" void kernel_launch(void* const* d_in, const int* in_sizes, int n_in,
                              void* d_out, int out_size, void* d_ws, size_t ws_size,
                              hipStream_t stream) {
  const float* x  = (const float*)d_in[0];
  const float* Wk = (const float*)d_in[1];
  const float* Wq = (const float*)d_in[2];
  const float* Wv = (const float*)d_in[3];
  float* out = (float*)d_out;

  char* ws = (char*)d_ws;
  bft* qb = (bft*)(ws);                                   // 2 MB  (BT,64) bf16
  bft* kb = (bft*)(ws + (size_t)2 * 1024 * 1024);         // 2 MB  (BT,64) bf16
  bft* vT = (bft*)(ws + (size_t)4 * 1024 * 1024);         // 2 MB  (64,BT) bf16
  bft* WT = (bft*)(ws + (size_t)6 * 1024 * 1024);         // 384 KB (3,64,1024) bf16

  wt_convert<<<(3 * HEADS * CDIM + 255) / 256, 256, 0, stream>>>(Wk, Wq, Wv, WT);
  qkv_proj<<<128, 256, 0, stream>>>(x, WT, kb, qb, vT);
  attn<<<128, 256, 0, stream>>>(qb, kb, vT, out);
}
